// CT_Transformer_27831388078461
// MI455X (gfx1250) — matrix operand order repaired and checked by reading, not compiled, not run on hardware
//
#include <hip/hip_runtime.h>
#include <hip/hip_bf16.h>

typedef __attribute__((ext_vector_type(16))) _Float16 v16h;
typedef __attribute__((ext_vector_type(8)))  _Float16 v8h;
typedef __attribute__((ext_vector_type(4)))  _Float16 v4h;
typedef __attribute__((ext_vector_type(8)))  float    v8f;

union AFrag { v16h v; v8h h[2]; };

__device__ __forceinline__ v8f wmma32(v16h a, v16h b, v8f c) {
  return __builtin_amdgcn_wmma_f32_16x16x32_f16(false, a, false, b, (short)0, c, false, false);
}
__device__ __forceinline__ float maxf(float a, float b) { return a > b ? a : b; }

// ---- CDNA5 async global->LDS staging (guarded; falls back to sync copies) ----
#if __has_builtin(__builtin_amdgcn_global_load_async_to_lds_b128)
#define HAVE_ASYNC_LDS 1
typedef int i32x4 __attribute__((vector_size(16)));          // matches builtin param
typedef __attribute__((address_space(1))) i32x4 gi32x4;      // global (AS1)
typedef __attribute__((address_space(3))) i32x4 li32x4;      // LDS (AS3)
__device__ __forceinline__ void stage32B(const _Float16* g, _Float16* l) {
  gi32x4* gp = (gi32x4*)(unsigned long long)g;
  li32x4* lp = (li32x4*)(unsigned int)(unsigned long long)l; // low 32b = LDS offset
  __builtin_amdgcn_global_load_async_to_lds_b128(gp, lp, 0, 0);
  __builtin_amdgcn_global_load_async_to_lds_b128(gp, lp, 16, 0);
}
__device__ __forceinline__ void stage_wait() {
#if __has_builtin(__builtin_amdgcn_s_wait_asynccnt)
  __builtin_amdgcn_s_wait_asynccnt(0);
#else
  asm volatile("s_wait_asynccnt 0" ::: "memory");
#endif
}
#else
#define HAVE_ASYNC_LDS 0
__device__ __forceinline__ void stage32B(const _Float16* g, _Float16* l) {
  const float4* s = (const float4*)g;
  float4* d = (float4*)l;
  d[0] = s[0]; d[1] = s[1];
}
__device__ __forceinline__ void stage_wait() {}
#endif

// ---------------------------------------------------------------- utilities
__global__ void __launch_bounds__(256) cvt_kernel(const float* __restrict__ s,
                                                  _Float16* __restrict__ d, int n) {
  int i = blockIdx.x * 256 + threadIdx.x;
  if (i < n) d[i] = (_Float16)s[i];
}

// Channel LayerNorm over dim=1 of NCHW (C=256, HW=4096). One thread per position.
__global__ void __launch_bounds__(256)
ln_kernel(const float* __restrict__ sx, const float* __restrict__ sy,
          const float* __restrict__ g, const float* __restrict__ bb,
          float* __restrict__ ox, float* __restrict__ oy,
          _Float16* __restrict__ hx, _Float16* __restrict__ hy) {
  const float* s = blockIdx.z ? sy : sx;
  float* o32 = blockIdx.z ? oy : ox;
  _Float16* o16 = blockIdx.z ? hy : hx;
  int pos = blockIdx.x * 256 + threadIdx.x;
  size_t base = (size_t)blockIdx.y * 256 * 4096 + pos;
  float sum = 0.f, sq = 0.f;
  for (int c = 0; c < 256; ++c) {
    float v = s[base + (size_t)c * 4096];
    sum += v; sq += v * v;
  }
  float mean = sum * (1.0f / 256.0f);
  float var  = sq  * (1.0f / 256.0f) - mean * mean;
  float rstd = rsqrtf(var + 1e-5f);
  for (int c = 0; c < 256; ++c) {
    float v = (s[base + (size_t)c * 4096] - mean) * rstd * g[c] + bb[c];
    if (o32) o32[base + (size_t)c * 4096] = v;
    if (o16) o16[base + (size_t)c * 4096] = (_Float16)v;
  }
}

// Depthwise 3x3 conv + folded eval-BatchNorm -> f16 [b][c][HoWo]. grid.y = stream.
__global__ void __launch_bounds__(256)
dw_kernel(const float* __restrict__ inx, const float* __restrict__ iny,
          const float* __restrict__ wdw, const float* __restrict__ bng,
          const float* __restrict__ bnb, _Float16* __restrict__ ox,
          _Float16* __restrict__ oy, int stride, int Ho, int Wo) {
  const float* in = blockIdx.y ? iny : inx;
  _Float16* out = blockIdx.y ? oy : ox;
  int idx = blockIdx.x * 256 + threadIdx.x;
  int wo = idx % Wo; int t = idx / Wo;
  int ho = t % Ho; t /= Ho;
  int c = t % 256; int b = t / 256;
  const float* p = in + ((size_t)b * 256 + c) * 4096;
  float acc = 0.f;
  for (int dy = 0; dy < 3; ++dy) {
    int iy = ho * stride + dy - 1;
    if (iy < 0 || iy >= 64) continue;
    for (int dx = 0; dx < 3; ++dx) {
      int ix = wo * stride + dx - 1;
      if (ix < 0 || ix >= 64) continue;
      acc += p[iy * 64 + ix] * wdw[c * 9 + dy * 3 + dx];
    }
  }
  float v = acc * (bng[c] * rsqrtf(1.0f + 1e-5f)) + bnb[c];
  out[((size_t)b * 256 + c) * (size_t)(Ho * Wo) + ho * Wo + wo] = (_Float16)v;
}

// Repack K: KV[b][512][1024] rows 0..255 (d-major) -> Kt[(b*4+h)*1024 + j][64] (j-major).
__global__ void __launch_bounds__(256)
krep_kernel(const _Float16* __restrict__ KVx, const _Float16* __restrict__ KVy,
            _Float16* __restrict__ Ktx, _Float16* __restrict__ Kty) {
  const _Float16* KV = blockIdx.y ? KVy : KVx;
  _Float16* Kt = blockIdx.y ? Kty : Ktx;
  int e = blockIdx.x * 256 + threadIdx.x;   // 4*4*64*1024 elements
  int j = e & 1023;
  int rest = e >> 10;
  int d = rest & 63;
  int bh = rest >> 6;                        // b*4 + h
  _Float16 v = KV[((size_t)(bh >> 2) * 512 + (bh & 3) * 64 + d) * 1024 + j];
  Kt[((size_t)bh * 1024 + j) * 64 + d] = v;
}

// --------------------------------------------------------- generic WMMA GEMM
// D[b] = A(MxK) * B[b](KxN)  (+bias[M]) (GELU) (+res[b]) -> out32/out16 (MxN)
// A f16 row-major weights (shared across batch); B f16 row-major per batch.
// grid = (N/64, M/64, batch), block = 128 (4 waves, each 32x32). N,K compile-time.
template <int N, int K>
__global__ void __launch_bounds__(128)
gemm_f16(const _Float16* __restrict__ A, const _Float16* __restrict__ Bg,
         const float* __restrict__ bias, const float* __restrict__ res,
         float* __restrict__ out32, _Float16* __restrict__ out16, int M, int act) {
  __shared__ __align__(16) _Float16 sA[64][32];   // row-major [m][k]
  __shared__ __align__(16) _Float16 sB[32][64];   // row-major [k][n]
  const int tid = threadIdx.x;
  const int lane = tid & 31, wid = tid >> 5;
  const int hi = lane >> 4, l15 = lane & 15;
  const int m0 = blockIdx.y * 64, n0 = blockIdx.x * 64;
  const size_t bz = blockIdx.z;
  const int wm = (wid >> 1) * 32, wn = (wid & 1) * 32;

  // hoisted per-thread staging pointers (advanced by constants each K-step)
  const _Float16* aSrc = A + (size_t)(m0 + (tid >> 1)) * K + (tid & 1) * 16;
  const _Float16* bSrc = Bg + bz * (size_t)K * N + (size_t)(tid >> 2) * N + n0 + (tid & 3) * 16;
  _Float16* aDst = &sA[tid >> 1][(tid & 1) * 16];
  _Float16* bDst = &sB[tid >> 2][(tid & 3) * 16];

  v8f acc[2][2] = {};
  for (int k0 = 0; k0 < K; k0 += 32) {
    stage32B(aSrc, aDst);
    stage32B(bSrc, bDst);
    stage_wait();
    __syncthreads();
    AFrag af[2]; v16h bf[2];
#pragma unroll
    for (int mi = 0; mi < 2; ++mi) {
      int m = wm + mi * 16 + l15;
      af[mi].h[0] = *(const v8h*)(&sA[m][hi * 16]);
      af[mi].h[1] = *(const v8h*)(&sA[m][8 + hi * 16]);
    }
#pragma unroll
    for (int ni = 0; ni < 2; ++ni) {
      const _Float16* col = &sB[16 * hi][wn + ni * 16 + l15];
      v16h b;
#pragma unroll
      for (int i = 0; i < 16; ++i) b[i] = col[i * 64];  // k = i + 16*hi
      bf[ni] = b;
    }
#pragma unroll
    for (int mi = 0; mi < 2; ++mi)
#pragma unroll
      for (int ni = 0; ni < 2; ++ni)
        acc[mi][ni] = wmma32(af[mi].v, bf[ni], acc[mi][ni]);
    __syncthreads();
    aSrc += 32;
    bSrc += (size_t)32 * N;
  }
  const size_t ob = bz * (size_t)M * N;
#pragma unroll
  for (int mi = 0; mi < 2; ++mi)
#pragma unroll
    for (int ni = 0; ni < 2; ++ni)
#pragma unroll
      for (int r = 0; r < 8; ++r) {
        int row = m0 + wm + mi * 16 + r + 8 * hi;
        int col = n0 + wn + ni * 16 + l15;
        float v = acc[mi][ni][r];
        if (bias) v += bias[row];
        if (act)  v = 0.5f * v * (1.0f + erff(v * 0.70710678118654752f));
        size_t o = ob + (size_t)row * N + col;
        if (res)   v += res[o];
        if (out32) out32[o] = v;
        if (out16) out16[o] = (_Float16)v;
      }
}

// ------------------------------------------------------------- attention
// One wave per (b, head, 16 q-rows). Q: [b][256][4096] f16 (d-major),
// KV: [b][512][1024] f16 (V rows 256..511, d-major), Kt: [(b*4+h)*1024+j][64] (j-major).
// O: [b][256][4096].
__global__ void __launch_bounds__(32)
attn_kernel(const _Float16* __restrict__ Q, const _Float16* __restrict__ KV,
            const _Float16* __restrict__ Kt, _Float16* __restrict__ O) {
  __shared__ __align__(16) float    sS[16 * 1024];   // 64 KB scores
  __shared__ __align__(16) _Float16 sP[16 * 1024];   // 32 KB probs
  __shared__ float sSum[16];
  const int lane = threadIdx.x, hi = lane >> 4, l15 = lane & 15;
  const int nq0 = blockIdx.x * 16, h = blockIdx.y, b = blockIdx.z;
  const _Float16* qb  = Q  + (size_t)b * 256 * 4096 + (size_t)h * 64 * 4096;
  const _Float16* vb  = KV + (size_t)b * 512 * 1024 + (size_t)(256 + h * 64) * 1024;
  const _Float16* ktb = Kt + (size_t)(b * 4 + h) * 1024 * 64;
  // Q A-fragments (K=64 contraction split into 2 x 32)
  AFrag aq[2];
#pragma unroll
  for (int s = 0; s < 2; ++s)
#pragma unroll
    for (int i = 0; i < 16; ++i) {
      int d = i + 16 * hi + 32 * s;
      aq[s].v[i] = qb[(size_t)d * 4096 + nq0 + l15];
    }
  // S = (Q K^T) / 8 : 64 column-tiles of 16; K fragments are contiguous b128 loads
  for (int jt = 0; jt < 64; ++jt) {
    const _Float16* kcol = ktb + (size_t)(jt * 16 + l15) * 64 + 16 * hi;
    v16h bk0 = *(const v16h*)(kcol);        // d = 16*hi + 0..15
    v16h bk1 = *(const v16h*)(kcol + 32);   // d = 32 + 16*hi + 0..15
    v8f c = {};
    c = wmma32(aq[0].v, bk0, c);
    c = wmma32(aq[1].v, bk1, c);
#pragma unroll
    for (int r = 0; r < 8; ++r)
      sS[(r + 8 * hi) * 1024 + jt * 16 + l15] = c[r] * 0.125f;
  }
  __syncthreads();
  { // softmax: lane handles row l15, half hi of 1024 cols (vectorized)
    const float4* row4 = (const float4*)(sS + l15 * 1024 + hi * 512);
    float mx = -3.4e38f;
    for (int j = 0; j < 128; ++j) {
      float4 v = row4[j];
      mx = maxf(mx, maxf(maxf(v.x, v.y), maxf(v.z, v.w)));
    }
    mx = maxf(mx, __shfl_xor(mx, 16, 32));
    float sm = 0.f;
    _Float16* prow = sP + l15 * 1024 + hi * 512;
    for (int j = 0; j < 128; ++j) {
      float4 v = row4[j];
      float p0 = __expf(v.x - mx), p1 = __expf(v.y - mx);
      float p2 = __expf(v.z - mx), p3 = __expf(v.w - mx);
      sm += (p0 + p1) + (p2 + p3);
      v4h ph = { (_Float16)p0, (_Float16)p1, (_Float16)p2, (_Float16)p3 };
      *(v4h*)(prow + 4 * j) = ph;
    }
    sm += __shfl_xor(sm, 16, 32);
    if (hi == 0) sSum[l15] = sm;
  }
  __syncthreads();
  // O = P V : K=1024 in 32 steps, 4 d-tiles of 16
  v8f co[4] = {};
  for (int kt = 0; kt < 32; ++kt) {
    AFrag ap;
    ap.h[0] = *(const v8h*)(&sP[l15 * 1024 + 32 * kt + 16 * hi]);
    ap.h[1] = *(const v8h*)(&sP[l15 * 1024 + 32 * kt + 8 + 16 * hi]);
#pragma unroll
    for (int dt = 0; dt < 4; ++dt) {
      v16h bv = *(const v16h*)(vb + (size_t)(dt * 16 + l15) * 1024 + 32 * kt + 16 * hi);
      co[dt] = wmma32(ap.v, bv, co[dt]);
    }
  }
  float inv[8];
#pragma unroll
  for (int r = 0; r < 8; ++r) inv[r] = 1.0f / sSum[r + 8 * hi];
  _Float16* ob = O + (size_t)b * 256 * 4096 + (size_t)h * 64 * 4096;
#pragma unroll
  for (int dt = 0; dt < 4; ++dt)
#pragma unroll
    for (int r = 0; r < 8; ++r)
      ob[(size_t)(dt * 16 + l15) * 4096 + nq0 + r + 8 * hi] = (_Float16)(co[dt][r] * inv[r]);
}

// ---------------------------------------------------------------- launcher
extern "C" void kernel_launch(void* const* d_in, const int* in_sizes, int n_in,
                              void* d_out, int out_size, void* d_ws, size_t ws_size,
                              hipStream_t stream) {
  (void)in_sizes; (void)n_in; (void)out_size; (void)ws_size;
  // Only the last layer (l=1) matters: the reference never feeds x4/y4 forward.
  const float* x      = (const float*)d_in[0];
  const float* y      = (const float*)d_in[1];
  const float* ng1    = (const float*)d_in[2]  + 256;
  const float* nb1    = (const float*)d_in[3]  + 256;
  const float* wq_dw  = (const float*)d_in[4]  + 256 * 9;
  const float* q_g    = (const float*)d_in[5]  + 256;
  const float* q_b    = (const float*)d_in[6]  + 256;
  const float* wq_pw  = (const float*)d_in[7]  + 256 * 256;
  const float* wkv_dw = (const float*)d_in[8]  + 256 * 9;
  const float* kv_g   = (const float*)d_in[9]  + 256;
  const float* kv_b   = (const float*)d_in[10] + 256;
  const float* wkv_pw = (const float*)d_in[11] + 512 * 256;
  const float* wo     = (const float*)d_in[12] + 256 * 256;
  const float* bo     = (const float*)d_in[13] + 256;
  const float* ng2    = (const float*)d_in[14] + 256;
  const float* nb2    = (const float*)d_in[15] + 256;
  const float* w1     = (const float*)d_in[16] + 1024 * 256;
  const float* b1     = (const float*)d_in[17] + 1024;
  const float* w2     = (const float*)d_in[18] + 256 * 1024;
  const float* b2     = (const float*)d_in[19] + 256;
  float* out = (float*)d_out;

  char* ws = (char*)d_ws;
  const size_t MB = 1ull << 20;
  // f16 weights [0, 2MB)
  _Float16* Wq  = (_Float16*)(ws + 0);
  _Float16* Wkv = (_Float16*)(ws + 128 * 1024);
  _Float16* Wo  = (_Float16*)(ws + 384 * 1024);
  _Float16* W1  = (_Float16*)(ws + 512 * 1024);
  _Float16* W2  = (_Float16*)(ws + 1024 * 1024);
  // activations (regions reused across sequential stages)
  float*    xn_x  = (float*)(ws + 2 * MB);      // 16MB
  float*    xn_y  = (float*)(ws + 18 * MB);     // 16MB
  _Float16* hq_x  = (_Float16*)(ws + 34 * MB);  // 8MB
  _Float16* hq_y  = (_Float16*)(ws + 42 * MB);  // 8MB
  _Float16* hkv_x = (_Float16*)(ws + 50 * MB);  // 2MB
  _Float16* hkv_y = (_Float16*)(ws + 52 * MB);  // 2MB
  _Float16* Q_x   = (_Float16*)(ws + 54 * MB);  // 8MB
  _Float16* Q_y   = (_Float16*)(ws + 62 * MB);  // 8MB
  _Float16* KV_x  = (_Float16*)(ws + 70 * MB);  // 4MB
  _Float16* KV_y  = (_Float16*)(ws + 74 * MB);  // 4MB
  _Float16* O_x   = (_Float16*)(ws + 34 * MB);  // reuse hq_x (hq dead)
  _Float16* O_y   = (_Float16*)(ws + 42 * MB);  // reuse hq_y
  _Float16* Kt_x  = (_Float16*)(ws + 50 * MB);  // reuse hkv_x (2MB)
  _Float16* Kt_y  = (_Float16*)(ws + 52 * MB);  // reuse hkv_y (2MB)
  float*    x2_x  = (float*)(ws + 78 * MB);     // 16MB
  float*    x2_y  = (float*)(ws + 94 * MB);     // 16MB
  _Float16* h2_x  = (_Float16*)(ws + 54 * MB);  // reuse Q_x (Q dead)
  _Float16* h2_y  = (_Float16*)(ws + 62 * MB);  // reuse Q_y
  _Float16* H1    = (_Float16*)(ws + 2 * MB);   // reuse xn region, 32MB

  // 1) weights -> f16
  cvt_kernel<<<256,  256, 0, stream>>>(wq_pw,  Wq,  65536);
  cvt_kernel<<<512,  256, 0, stream>>>(wkv_pw, Wkv, 131072);
  cvt_kernel<<<256,  256, 0, stream>>>(wo,     Wo,  65536);
  cvt_kernel<<<1024, 256, 0, stream>>>(w1,     W1,  262144);
  cvt_kernel<<<1024, 256, 0, stream>>>(w2,     W2,  262144);

  // 2) LayerNorm 1 (f32 out)
  ln_kernel<<<dim3(16, 4, 2), 256, 0, stream>>>(x, y, ng1, nb1, xn_x, xn_y,
                                                (_Float16*)nullptr, (_Float16*)nullptr);
  // 3) depthwise convs + BN -> f16
  dw_kernel<<<dim3(16384, 2), 256, 0, stream>>>(xn_x, xn_y, wq_dw,  q_g,  q_b,
                                                hq_x,  hq_y,  1, 64, 64);
  dw_kernel<<<dim3(4096,  2), 256, 0, stream>>>(xn_x, xn_y, wkv_dw, kv_g, kv_b,
                                                hkv_x, hkv_y, 2, 32, 32);
  // 4) pointwise projections (WMMA GEMMs)
  gemm_f16<4096, 256><<<dim3(64, 4, 4), 128, 0, stream>>>(Wq,  hq_x,  nullptr, nullptr, nullptr, Q_x, 256, 0);
  gemm_f16<4096, 256><<<dim3(64, 4, 4), 128, 0, stream>>>(Wq,  hq_y,  nullptr, nullptr, nullptr, Q_y, 256, 0);
  gemm_f16<1024, 256><<<dim3(16, 8, 4), 128, 0, stream>>>(Wkv, hkv_x, nullptr, nullptr, nullptr, KV_x, 512, 0);
  gemm_f16<1024, 256><<<dim3(16, 8, 4), 128, 0, stream>>>(Wkv, hkv_y, nullptr, nullptr, nullptr, KV_y, 512, 0);
  // 4b) repack K j-major for contiguous WMMA B-fragment loads
  krep_kernel<<<dim3(4096, 2), 256, 0, stream>>>(KV_x, KV_y, Kt_x, Kt_y);
  // 5) cross attention: ox = attn(qx, ky, vy), oy = attn(qy, kx, vx)
  attn_kernel<<<dim3(256, 4, 4), 32, 0, stream>>>(Q_x, KV_y, Kt_y, O_x);
  attn_kernel<<<dim3(256, 4, 4), 32, 0, stream>>>(Q_y, KV_x, Kt_x, O_y);
  // 6) output projection + residual
  gemm_f16<4096, 256><<<dim3(64, 4, 4), 128, 0, stream>>>(Wo, O_x, bo, x, x2_x, (_Float16*)nullptr, 256, 0);
  gemm_f16<4096, 256><<<dim3(64, 4, 4), 128, 0, stream>>>(Wo, O_y, bo, y, x2_y, (_Float16*)nullptr, 256, 0);
  // 7) LayerNorm 2 -> f16
  ln_kernel<<<dim3(16, 4, 2), 256, 0, stream>>>(x2_x, x2_y, ng2, nb2,
                                                (float*)nullptr, (float*)nullptr, h2_x, h2_y);
  // 8) FFN, x stream then y stream (H1 reused; stream serializes launches)
  gemm_f16<4096, 256><<<dim3(64, 16, 4), 128, 0, stream>>>(W1, h2_x, b1, nullptr, nullptr, H1, 1024, 1);
  gemm_f16<4096, 1024><<<dim3(64, 4, 4), 128, 0, stream>>>(W2, H1, b2, x2_x, out, (_Float16*)nullptr, 256, 0);
  gemm_f16<4096, 256><<<dim3(64, 16, 4), 128, 0, stream>>>(W1, h2_y, b1, nullptr, nullptr, H1, 1024, 1);
  gemm_f16<4096, 1024><<<dim3(64, 4, 4), 128, 0, stream>>>(W2, H1, b2, x2_y, out + 4ull * 256 * 4096,
                                                           (_Float16*)nullptr, 256, 0);
}
